// HybridBlockRef_65146063946244
// MI455X (gfx1250) — matrix operand order repaired and checked by reading, not compiled, not run on hardware
//
#include <hip/hip_runtime.h>

#define B_ 8
#define T_ 1024
#define D_ 1024
#define H_ 4096
#define M_ (B_ * T_)   // 8192 rows

#if defined(__AMDGCN__) && __has_builtin(__builtin_amdgcn_global_load_async_to_lds_b128) && \
    __has_builtin(__builtin_amdgcn_s_wait_asynccnt)
#define USE_ASYNC 1
#else
#define USE_ASYNC 0
#endif

typedef __attribute__((ext_vector_type(4)))  _Float16 v4h;
typedef __attribute__((ext_vector_type(8)))  _Float16 v8h;
typedef __attribute__((ext_vector_type(16))) _Float16 v16h;
typedef __attribute__((ext_vector_type(8)))  float    v8f;
typedef __attribute__((ext_vector_type(4)))  int      v4i;

static __device__ __forceinline__ v16h cat16(v8h lo, v8h hi) {
  return __builtin_shufflevector(lo, hi, 0, 1, 2, 3, 4, 5, 6, 7,
                                 8, 9, 10, 11, 12, 13, 14, 15);
}

#if USE_ASYNC
// address-space qualified int4 pointers per the builtin's signature
// (param0 diagnosed as "int __vector(4) __device__ *" == int4 addrspace(1)*)
typedef __attribute__((address_space(1))) v4i gv4i;
typedef __attribute__((address_space(3))) v4i lv4i;
// one wave-level async DMA of 16B/lane straight into LDS (ASYNCcnt-tracked)
static __device__ __forceinline__ void cp_async16(const _Float16* g, _Float16* l) {
  __builtin_amdgcn_global_load_async_to_lds_b128((gv4i*)g, (lv4i*)l, 0, 0);
}
#endif

// ---------------- fp32 -> fp16 convert (4 elems / thread) ----------------
__global__ __launch_bounds__(256) void k_f32_to_f16(const float* __restrict__ in,
                                                    _Float16* __restrict__ out,
                                                    int n4) {
  int i = blockIdx.x * 256 + threadIdx.x;
  if (i >= n4) return;
  float4 v = ((const float4*)in)[i];
  v4h o = { (_Float16)v.x, (_Float16)v.y, (_Float16)v.z, (_Float16)v.w };
  ((v4h*)out)[i] = o;
}

// ---------------- rmsnorm: one 256-thread block per row of D=1024 ----------------
__global__ __launch_bounds__(256) void k_rmsnorm_f16(const float* __restrict__ x,
                                                     const float* __restrict__ w,
                                                     _Float16* __restrict__ out) {
  __shared__ float red[256];
  const int row = blockIdx.x, tid = threadIdx.x;
  float4 xv = ((const float4*)(x + (size_t)row * D_))[tid];
  red[tid] = xv.x * xv.x + xv.y * xv.y + xv.z * xv.z + xv.w * xv.w;
  __syncthreads();
  for (int s = 128; s > 0; s >>= 1) {
    if (tid < s) red[tid] += red[tid + s];
    __syncthreads();
  }
  float scale = rsqrtf(red[0] * (1.0f / (float)D_) + 1e-6f);
  float4 wv = ((const float4*)w)[tid];
  v4h o = { (_Float16)(xv.x * wv.x * scale), (_Float16)(xv.y * wv.y * scale),
            (_Float16)(xv.z * wv.z * scale), (_Float16)(xv.w * wv.w * scale) };
  ((v4h*)(out + (size_t)row * D_))[tid] = o;
}

// ---------------- epilogues ----------------
struct EpiCfc {           // delta = softplus(g0); At = exp(-delta*exp(A_log)); Bt = delta*g1
  const float* A_log; float* At; float* Bt;
  __device__ void operator()(int row, int col, float g0, float g1) const {
    float delta = (g0 > 20.f) ? g0 : log1pf(__expf(g0));
    size_t i = (size_t)row * D_ + col;
    At[i] = __expf(-delta * __expf(A_log[col]));
    Bt[i] = delta * g1;
  }
};
struct EpiSiluUp {        // act = silu(gate) * up  -> f16
  _Float16* act;
  __device__ void operator()(int row, int col, float g, float u) const {
    float s = g / (1.f + __expf(-g));
    act[(size_t)row * H_ + col] = (_Float16)(s * u);
  }
};
struct EpiResAdd {        // y = x + acc  (full overwrite of d_out)
  const float* x; float* y;
  __device__ void operator()(int row, int col, float g0, float) const {
    size_t i = (size_t)row * D_ + col;
    y[i] = x[i] + g0;
  }
};
struct EpiAccum {         // y += acc  (ffn residual into d_out)
  float* y;
  __device__ void operator()(int row, int col, float g0, float) const {
    size_t i = (size_t)row * D_ + col;
    y[i] += g0;
  }
};

// ---------------- WMMA GEMM: C[M,N] = A[M,K] x W[N,K]^T, optional dual W ----------------
// Block tile 128(M) x 128(N), 8 wave32s (2x4), wave tile 64x32, K-step 32.
// LDS double-buffered; filled by global_load_async_to_lds_b128 (ASYNCcnt) when
// available, else by a register-pipelined copy with distinct temps.
template <bool DUAL, class Epi>
__global__ __launch_bounds__(256) void k_gemm_wmma(const _Float16* __restrict__ A,
                                                   const _Float16* __restrict__ W0,
                                                   const _Float16* __restrict__ W1,
                                                   int K, Epi epi) {
  __shared__ _Float16 As[2][128][40];
  __shared__ _Float16 Bs[2][DUAL ? 2 : 1][128][40];
  const int tid  = threadIdx.x;
  const int lane = tid & 31;
  const int wave = tid >> 5;
  const int l16  = lane & 15;
  const int hi   = lane >> 4;            // half-wave id
  const int wm   = (wave & 1) * 64;      // wave M offset in block tile
  const int wn   = (wave >> 1) * 32;     // wave N offset in block tile
  const int bm   = blockIdx.y * 128;
  const int bn   = blockIdx.x * 128;

  // tile loader: 128 rows x 32 cols, 16 halves (2 x b128) per thread
  const int arow = tid >> 1, acol = (tid & 1) * 16;

  v8f acc0[4][2] = {};
  v8f acc1[4][2] = {};

  const _Float16* gA  = A + (size_t)(bm + arow) * K + acol;
  const _Float16* gW0 = W0 + (size_t)(bn + arow) * K + acol;
  const _Float16* gW1 = DUAL ? (W1 + (size_t)(bn + arow) * K + acol) : gW0;

  const int nk = K >> 5;
  constexpr int NLOADS = DUAL ? 6 : 4;   // async-load instrs per wave per K-step
  (void)NLOADS;

#if USE_ASYNC
  auto issue = [&](int k0, int buf) {
    cp_async16(gA + k0,      &As[buf][arow][acol]);
    cp_async16(gA + k0 + 8,  &As[buf][arow][acol + 8]);
    cp_async16(gW0 + k0,     &Bs[buf][0][arow][acol]);
    cp_async16(gW0 + k0 + 8, &Bs[buf][0][arow][acol + 8]);
    if constexpr (DUAL) {
      cp_async16(gW1 + k0,     &Bs[buf][1][arow][acol]);
      cp_async16(gW1 + k0 + 8, &Bs[buf][1][arow][acol + 8]);
    }
  };
  issue(0, 0);
#else
  v8h ra0, ra1, rb0, rb1, rc0, rc1;
  auto fetch = [&](int k0) {             // distinct temps -> overlapped loads
    ra0 = *(const v8h*)(gA + k0);
    ra1 = *(const v8h*)(gA + k0 + 8);
    rb0 = *(const v8h*)(gW0 + k0);
    rb1 = *(const v8h*)(gW0 + k0 + 8);
    if constexpr (DUAL) {
      rc0 = *(const v8h*)(gW1 + k0);
      rc1 = *(const v8h*)(gW1 + k0 + 8);
    }
  };
  auto stage = [&](int buf) {
    *(v8h*)&As[buf][arow][acol]        = ra0;
    *(v8h*)&As[buf][arow][acol + 8]    = ra1;
    *(v8h*)&Bs[buf][0][arow][acol]     = rb0;
    *(v8h*)&Bs[buf][0][arow][acol + 8] = rb1;
    if constexpr (DUAL) {
      *(v8h*)&Bs[buf][1][arow][acol]     = rc0;
      *(v8h*)&Bs[buf][1][arow][acol + 8] = rc1;
    }
  };
  fetch(0);
#endif

  for (int kt = 0; kt < nk; ++kt) {
    const int buf = kt & 1;
#if USE_ASYNC
    if (kt + 1 < nk) {
      issue((kt + 1) << 5, buf ^ 1);     // stream next tile while computing
      __builtin_amdgcn_s_wait_asynccnt(NLOADS);  // oldest set (this buf) done
    } else {
      __builtin_amdgcn_s_wait_asynccnt(0);
    }
    __syncthreads();                     // all waves' fills of `buf` visible
#else
    stage(buf);
    __syncthreads();
    if (kt + 1 < nk) fetch((kt + 1) << 5);
#endif

    // B fragments: lanes 0-15 hold K 0..15, lanes 16-31 hold K 16..31
    v16h bf0[2], bf1[2];
#pragma unroll
    for (int n = 0; n < 2; ++n) {
      bf0[n] = cat16(*(const v8h*)&Bs[buf][0][wn + n * 16 + l16][hi * 16],
                     *(const v8h*)&Bs[buf][0][wn + n * 16 + l16][hi * 16 + 8]);
      if constexpr (DUAL)
        bf1[n] = cat16(*(const v8h*)&Bs[buf][1][wn + n * 16 + l16][hi * 16],
                       *(const v8h*)&Bs[buf][1][wn + n * 16 + l16][hi * 16 + 8]);
    }
#pragma unroll
    for (int s = 0; s < 4; ++s) {
      // A fragment: lanes 0-15 hold K 0..15, lanes 16-31 hold K 16..31
      v16h af = cat16(*(const v8h*)&As[buf][wm + s * 16 + l16][hi * 16],
                      *(const v8h*)&As[buf][wm + s * 16 + l16][hi * 16 + 8]);
#pragma unroll
      for (int n = 0; n < 2; ++n) {
        acc0[s][n] = __builtin_amdgcn_wmma_f32_16x16x32_f16(false, af, false, bf0[n],
                                                            (short)0, acc0[s][n], false, false);
        if constexpr (DUAL)
          acc1[s][n] = __builtin_amdgcn_wmma_f32_16x16x32_f16(false, af, false, bf1[n],
                                                              (short)0, acc1[s][n], false, false);
      }
    }
    __syncthreads();                     // reads of `buf` done before it refills
  }

  // C layout: VGPR r -> row r (lanes 0-15) / r+8 (lanes 16-31), col = lane%16
#pragma unroll
  for (int s = 0; s < 4; ++s) {
#pragma unroll
    for (int n = 0; n < 2; ++n) {
      const int row = bm + wm + s * 16 + hi * 8;
      const int col = bn + wn + n * 16 + l16;
#pragma unroll
      for (int r = 0; r < 8; ++r)
        epi(row + r, col, acc0[s][n][r], acc1[s][n][r]);
    }
  }
}

// ---------------- fused CfC + PLIF sequential scan over T ----------------
__global__ __launch_bounds__(256) void k_scan(const float* __restrict__ At,
                                              const float* __restrict__ Bt,
                                              const float* __restrict__ tau_log,
                                              const float* __restrict__ thr,
                                              _Float16* __restrict__ s_out) {
  int i = blockIdx.x * 256 + threadIdx.x;   // exactly B_*D_ = 8192 threads
  int d = i & (D_ - 1);
  int b = i >> 10;
  float decay = __expf(-1.f / __expf(tau_log[d]));
  float th = thr[d];
  size_t off = (size_t)b * T_ * D_ + d;
  float h = 0.f, mem = 0.f;
  float a = At[off], bb = Bt[off];          // software-pipelined loads
  for (int t = 0; t < T_; ++t) {
    size_t nxt = off + D_;
    float a_n = 0.f, b_n = 0.f;
    if (t + 1 < T_) { a_n = At[nxt]; b_n = Bt[nxt]; }
    h = fmaf(a, h, bb);                     // h = A_t*h + B_t
    float c = tanhf(h);
    mem = fmaf(mem, decay, c);              // mem = mem*decay + c
    float sp = (mem >= th) ? 1.f : 0.f;
    mem -= sp * th;
    s_out[off] = (_Float16)sp;
    a = a_n; bb = b_n; off = nxt;
  }
}

extern "C" void kernel_launch(void* const* d_in, const int* in_sizes, int n_in,
                              void* d_out, int out_size, void* d_ws, size_t ws_size,
                              hipStream_t stream) {
  (void)in_sizes; (void)n_in; (void)out_size; (void)ws_size;
  const float* x       = (const float*)d_in[0];
  const float* ln1_w   = (const float*)d_in[1];
  const float* W_delta = (const float*)d_in[2];
  const float* W_b     = (const float*)d_in[3];
  const float* A_log   = (const float*)d_in[4];
  const float* tau_log = (const float*)d_in[5];
  const float* thresh  = (const float*)d_in[6];
  const float* syn_w   = (const float*)d_in[7];
  const float* ln2_w   = (const float*)d_in[8];
  const float* w_gate  = (const float*)d_in[9];
  const float* w_up    = (const float*)d_in[10];
  const float* w_down  = (const float*)d_in[11];
  float* out = (float*)d_out;

  char* base = (char*)d_ws;
  const size_t MB = 1024ull * 1024ull;
  _Float16* a_h   = (_Float16*)(base + 0);        // 16MB; reused later as y_h
  _Float16* wd_h  = (_Float16*)(base + 16 * MB);  // 2MB
  _Float16* wb_h  = (_Float16*)(base + 18 * MB);  // 2MB
  _Float16* syn_h = (_Float16*)(base + 20 * MB);  // 2MB
  _Float16* wg_h  = (_Float16*)(base + 22 * MB);  // 8MB
  _Float16* wu_h  = (_Float16*)(base + 30 * MB);  // 8MB
  _Float16* wdn_h = (_Float16*)(base + 38 * MB);  // 8MB
  float*    At    = (float*)(base + 46 * MB);     // 32MB
  float*    Bt    = (float*)(base + 78 * MB);     // 32MB
  _Float16* act_h = (_Float16*)(base + 46 * MB);  // 64MB, recycles At+Bt after scan
  _Float16* s_h   = (_Float16*)(base + 110 * MB); // 16MB   (total ws use: 126MB)

  // 1) weight conversions fp32 -> fp16
  k_f32_to_f16<<<(D_ * D_ / 4) / 256, 256, 0, stream>>>(W_delta, wd_h, D_ * D_ / 4);
  k_f32_to_f16<<<(D_ * D_ / 4) / 256, 256, 0, stream>>>(W_b, wb_h, D_ * D_ / 4);
  k_f32_to_f16<<<(D_ * D_ / 4) / 256, 256, 0, stream>>>(syn_w, syn_h, D_ * D_ / 4);
  k_f32_to_f16<<<(H_ * D_ / 4) / 256, 256, 0, stream>>>(w_gate, wg_h, H_ * D_ / 4);
  k_f32_to_f16<<<(H_ * D_ / 4) / 256, 256, 0, stream>>>(w_up, wu_h, H_ * D_ / 4);
  k_f32_to_f16<<<(H_ * D_ / 4) / 256, 256, 0, stream>>>(w_down, wdn_h, H_ * D_ / 4);

  // 2) rmsnorm1 -> a (f16)
  k_rmsnorm_f16<<<M_, 256, 0, stream>>>(x, ln1_w, a_h);

  // 3) dual GEMM: delta-lin & b-lin, fused gate epilogue -> At, Bt (fp32)
  dim3 gD(D_ / 128, M_ / 128);  // (8, 64)
  k_gemm_wmma<true, EpiCfc><<<gD, 256, 0, stream>>>(a_h, wd_h, wb_h, D_,
                                                    EpiCfc{A_log, At, Bt});

  // 4) sequential CfC + PLIF scan -> spikes (f16)
  k_scan<<<(B_ * D_) / 256, 256, 0, stream>>>(At, Bt, tau_log, thresh, s_h);

  // 5) x2 = x + s @ syn^T  (writes every element of d_out)
  k_gemm_wmma<false, EpiResAdd><<<gD, 256, 0, stream>>>(s_h, syn_h, nullptr, D_,
                                                        EpiResAdd{x, out});

  // 6) rmsnorm2 -> y (f16, reuse a_h)
  k_rmsnorm_f16<<<M_, 256, 0, stream>>>(out, ln2_w, a_h);

  // 7) dual GEMM gate/up + silu-mul -> act (f16, recycled At/Bt region)
  dim3 gH(H_ / 128, M_ / 128);  // (32, 64)
  k_gemm_wmma<true, EpiSiluUp><<<gH, 256, 0, stream>>>(a_h, wg_h, wu_h, D_,
                                                       EpiSiluUp{act_h});

  // 8) d_out += act @ w_down^T  (K = 4096)
  k_gemm_wmma<false, EpiAccum><<<gD, 256, 0, stream>>>(act_h, wdn_h, nullptr, H_,
                                                       EpiAccum{out});
}
